// MultiHeadSelfAttention_5463198400993
// MI455X (gfx1250) — matrix operand order repaired and checked by reading, not compiled, not run on hardware
//
#include <hip/hip_runtime.h>

#define D_MODEL 1024
#define SEQ     2048
#define BATCH   2
#define NH      16
#define HD      64
#define NROWS   (BATCH * SEQ)   // 4096

typedef unsigned short u16t;
typedef __attribute__((ext_vector_type(16))) __bf16 v16bf;
typedef __attribute__((ext_vector_type(8)))  float  v8f;

// ---------- helpers ----------
__device__ __forceinline__ u16t f2bf(float f) {
    union { float f; unsigned u; } x; x.f = f;
    unsigned u = x.u;
    unsigned r = u + 0x7FFFu + ((u >> 16) & 1u);   // RNE
    return (u16t)(r >> 16);
}

// A-matrix 16x32 bf16 fragment (per ISA table): lane L holds row (L&15),
// K chunks at (L>>4)*8 .. +7 and +16 .. +23  -> two 16B loads.
__device__ __forceinline__ v16bf load_frag_a(const u16t* p, int ld) {
    const int L  = threadIdx.x & 31;
    const int r  = L & 15;
    const int k0 = (L >> 4) << 3;
    const u16t* q = p + (size_t)r * ld + k0;
    union { v16bf v; uint4 u[2]; } f;
    f.u[0] = *reinterpret_cast<const uint4*>(q);
    f.u[1] = *reinterpret_cast<const uint4*>(q + 16);
    return f.v;
}

// B-matrix 32x16 bf16 fragment: lane L holds column (L&15),
// K chunks at (L>>4)*8 .. +7 and +16 .. +23  -> two 16B loads.
__device__ __forceinline__ v16bf load_frag_b(const u16t* p, int ld) {
    const int L  = threadIdx.x & 31;
    const int n  = L & 15;
    const int k0 = (L >> 4) << 3;
    const u16t* q = p + (size_t)n * ld + k0;
    union { v16bf v; uint4 u[2]; } f;
    f.u[0] = *reinterpret_cast<const uint4*>(q);
    f.u[1] = *reinterpret_cast<const uint4*>(q + 16);
    return f.v;
}

__device__ __forceinline__ v8f wmma_bf(v16bf a, v16bf b, v8f c) {
    return __builtin_amdgcn_wmma_f32_16x16x32_bf16(false, a, false, b,
                                                   (short)0, c, false, false);
}

// ---------- f32 -> bf16 conversion ----------
__global__ void cvt_f32_to_bf16(const float4* __restrict__ src,
                                ushort4* __restrict__ dst, int n4) {
    int i = blockIdx.x * blockDim.x + threadIdx.x;
    if (i < n4) {
        float4 f = src[i];
        ushort4 o;
        o.x = f2bf(f.x); o.y = f2bf(f.y); o.z = f2bf(f.z); o.w = f2bf(f.w);
        dst[i] = o;
    }
}

// ---------- fused QKV projection + RoPE ----------
// grid (32, 8, 3): x = M tiles of 128 (4096 rows), y = N tiles of 128, z = {Q,K,V}
__global__ __launch_bounds__(256) void qkv_rope_kernel(
    const u16t* __restrict__ xb,
    const u16t* __restrict__ wq, const u16t* __restrict__ wk,
    const u16t* __restrict__ wv,
    u16t* __restrict__ Qr, u16t* __restrict__ Kr, u16t* __restrict__ Vt)
{
    const int which = blockIdx.z;
    const u16t* W = (which == 0) ? wq : (which == 1) ? wk : wv;
    const int wid  = threadIdx.x >> 5;
    const int L    = threadIdx.x & 31;
    const int half = L >> 4;
    const int nl   = L & 15;
    const int row0 = blockIdx.x * 128 + (wid & 3) * 32;   // 4 waves along M
    const int col0 = blockIdx.y * 128 + (wid >> 2) * 64;  // 2 waves along N

    v8f acc[2][4] = {};
    for (int k = 0; k < D_MODEL; k += 32) {
        v16bf a0 = load_frag_a(xb + (size_t)row0        * D_MODEL + k, D_MODEL);
        v16bf a1 = load_frag_a(xb + (size_t)(row0 + 16) * D_MODEL + k, D_MODEL);
        #pragma unroll
        for (int j = 0; j < 4; ++j) {
            v16bf b = load_frag_b(W + (size_t)(col0 + j * 16) * D_MODEL + k, D_MODEL);
            acc[0][j] = wmma_bf(a0, b, acc[0][j]);
            acc[1][j] = wmma_bf(a1, b, acc[1][j]);
        }
    }

    // epilogue: RoPE (Q,K) and scatter to [b,h,s,d]; V scattered transposed [b,h,d,s]
    #pragma unroll
    for (int j = 0; j < 4; ++j) {
        const int colg = col0 + j * 16 + nl;
        const int h = colg >> 6;
        const int d = colg & 63;
        // theta^(-2i/64) = exp(-(2i) * ln(1e4)/64), 2i = d&~1
        const float invf = __expf(-(float)(d & ~1) * (9.2103403719761836f / 64.0f));
        #pragma unroll
        for (int i = 0; i < 2; ++i) {
            #pragma unroll
            for (int v = 0; v < 8; ++v) {
                const int rowg = row0 + i * 16 + v + 8 * half;
                const int b = rowg >> 11;
                const int s = rowg & (SEQ - 1);
                float val = acc[i][j][v];
                if (which < 2) {
                    const float ang = (float)s * invf;
                    const float c = __cosf(ang), sn = __sinf(ang);
                    const float pv = __shfl_xor(val, 1, 32);  // partner column
                    val = (d & 1) ? (pv * sn + val * c) : (val * c - pv * sn);
                    u16t* dst = (which == 0) ? Qr : Kr;
                    dst[((size_t)(b * NH + h) * SEQ + s) * HD + d] = f2bf(val);
                } else {
                    Vt[((size_t)(b * NH + h) * HD + d) * SEQ + s] = f2bf(val);
                }
            }
        }
    }
}

// ---------- causal flash attention ----------
// grid (32, 16): x = b*NH+h, y = 128-row query block; wave owns 16 q rows.
__global__ __launch_bounds__(256) void attn_kernel(
    const u16t* __restrict__ Qr, const u16t* __restrict__ Kr,
    const u16t* __restrict__ Vt, u16t* __restrict__ AO)
{
    __shared__ __align__(16) u16t plds[8][16 * 32];   // per-wave P tile (16x32 bf16)
    const int wid  = threadIdx.x >> 5;
    const int L    = threadIdx.x & 31;
    const int half = L >> 4;
    const int nl   = L & 15;
    const int bh   = blockIdx.x;
    const int qr0  = blockIdx.y * 128 + wid * 16;

    const u16t* Qb = Qr + (size_t)bh * SEQ * HD;
    const u16t* Kb = Kr + (size_t)bh * SEQ * HD;
    const u16t* Vb = Vt + (size_t)bh * HD * SEQ;

    const v16bf qf0 = load_frag_a(Qb + (size_t)qr0 * HD,      HD);
    const v16bf qf1 = load_frag_a(Qb + (size_t)qr0 * HD + 32, HD);

    v8f o[4] = {};
    float mrow[8], lrow[8];
    #pragma unroll
    for (int v = 0; v < 8; ++v) { mrow[v] = -1e30f; lrow[v] = 0.0f; }

    const float scale = 0.125f;          // 1/sqrt(64)
    const int nkv = qr0 + 16;            // keys needed (multiple of 16)

    for (int kb = 0; kb < nkv; kb += 32) {
        const bool t1 = (kb + 16) < nkv; // second 16-key tile valid?
        v8f c0 = {}, c1 = {};
        {
            v16bf b0 = load_frag_b(Kb + (size_t)kb * HD,      HD);
            v16bf b1 = load_frag_b(Kb + (size_t)kb * HD + 32, HD);
            c0 = wmma_bf(qf0, b0, c0);
            c0 = wmma_bf(qf1, b1, c0);
        }
        if (t1) {
            v16bf b0 = load_frag_b(Kb + (size_t)(kb + 16) * HD,      HD);
            v16bf b1 = load_frag_b(Kb + (size_t)(kb + 16) * HD + 32, HD);
            c1 = wmma_bf(qf0, b0, c1);
            c1 = wmma_bf(qf1, b1, c1);
        }
        // online softmax; C-layout row = v + 8*half, col = lane in half
        float corr[8];
        #pragma unroll
        for (int v = 0; v < 8; ++v) {
            const int qg = qr0 + v + 8 * half;
            float s0 = c0[v] * scale;
            if (kb + nl > qg) s0 = -1e30f;
            float s1 = t1 ? (c1[v] * scale) : -1e30f;
            if (kb + 16 + nl > qg) s1 = -1e30f;
            float t = fmaxf(s0, s1);
            #pragma unroll
            for (int w = 1; w < 16; w <<= 1) t = fmaxf(t, __shfl_xor(t, w, 32));
            const float mnew = fmaxf(mrow[v], t);
            corr[v] = __expf(mrow[v] - mnew);
            mrow[v] = mnew;
            const float p0 = __expf(s0 - mnew);
            const float p1 = __expf(s1 - mnew);
            float rs = p0 + p1;
            #pragma unroll
            for (int w = 1; w < 16; w <<= 1) rs += __shfl_xor(rs, w, 32);
            lrow[v] = lrow[v] * corr[v] + rs;
            plds[wid][(v + 8 * half) * 32 + nl]      = f2bf(p0);
            plds[wid][(v + 8 * half) * 32 + 16 + nl] = f2bf(p1);
        }
        #pragma unroll
        for (int t4 = 0; t4 < 4; ++t4)
            #pragma unroll
            for (int v = 0; v < 8; ++v) o[t4][v] *= corr[v];

        asm volatile("s_wait_dscnt 0" ::: "memory");  // wave-internal LDS RAW

        v16bf pa;  // re-read P in A-fragment layout (16x32, ld=32)
        {
            const u16t* q = &plds[wid][(L & 15) * 32 + ((L >> 4) << 3)];
            union { v16bf v; uint4 u[2]; } f;
            f.u[0] = *reinterpret_cast<const uint4*>(q);
            f.u[1] = *reinterpret_cast<const uint4*>(q + 16);
            pa = f.v;
        }
        #pragma unroll
        for (int t4 = 0; t4 < 4; ++t4) {
            v16bf vb = load_frag_b(Vb + (size_t)(t4 * 16) * SEQ + kb, SEQ);
            o[t4] = wmma_bf(pa, vb, o[t4]);
        }
    }

    // normalize and store AO[b*S+s][h*64+d] (bf16)
    const int b = bh / NH;
    const int h = bh % NH;
    #pragma unroll
    for (int v = 0; v < 8; ++v) {
        const float inv = 1.0f / lrow[v];
        const int q = qr0 + v + 8 * half;
        const size_t rowbase = ((size_t)b * SEQ + q) * D_MODEL + h * HD;
        #pragma unroll
        for (int t4 = 0; t4 < 4; ++t4)
            AO[rowbase + t4 * 16 + nl] = f2bf(o[t4][v] * inv);
    }
}

// ---------- output projection: out = AO @ Wo^T (f32 out) ----------
__global__ __launch_bounds__(256) void oproj_kernel(
    const u16t* __restrict__ AO, const u16t* __restrict__ Wo,
    float* __restrict__ out)
{
    const int wid  = threadIdx.x >> 5;
    const int L    = threadIdx.x & 31;
    const int half = L >> 4;
    const int nl   = L & 15;
    const int row0 = blockIdx.x * 128 + (wid & 3) * 32;
    const int col0 = blockIdx.y * 128 + (wid >> 2) * 64;

    v8f acc[2][4] = {};
    for (int k = 0; k < D_MODEL; k += 32) {
        v16bf a0 = load_frag_a(AO + (size_t)row0        * D_MODEL + k, D_MODEL);
        v16bf a1 = load_frag_a(AO + (size_t)(row0 + 16) * D_MODEL + k, D_MODEL);
        #pragma unroll
        for (int j = 0; j < 4; ++j) {
            v16bf b = load_frag_b(Wo + (size_t)(col0 + j * 16) * D_MODEL + k, D_MODEL);
            acc[0][j] = wmma_bf(a0, b, acc[0][j]);
            acc[1][j] = wmma_bf(a1, b, acc[1][j]);
        }
    }
    #pragma unroll
    for (int i = 0; i < 2; ++i)
        #pragma unroll
        for (int j = 0; j < 4; ++j)
            #pragma unroll
            for (int v = 0; v < 8; ++v)
                out[(size_t)(row0 + i * 16 + v + 8 * half) * D_MODEL +
                    (col0 + j * 16 + nl)] = acc[i][j][v];
}

// ---------- host orchestration ----------
extern "C" void kernel_launch(void* const* d_in, const int* in_sizes, int n_in,
                              void* d_out, int out_size, void* d_ws, size_t ws_size,
                              hipStream_t stream)
{
    (void)in_sizes; (void)n_in; (void)out_size; (void)ws_size;
    const float* x  = (const float*)d_in[0];
    const float* Wq = (const float*)d_in[1];
    const float* Wk = (const float*)d_in[2];
    const float* Wv = (const float*)d_in[3];
    const float* Wo = (const float*)d_in[4];
    float* out = (float*)d_out;

    char* ws = (char*)d_ws;
    size_t off = 0;
    auto carve = [&](size_t bytes) -> char* {
        char* p = ws + off;
        off += (bytes + 255) & ~(size_t)255;
        return p;
    };
    u16t* xb  = (u16t*)carve((size_t)NROWS * D_MODEL * 2);     // 8 MiB
    u16t* wqb = (u16t*)carve((size_t)D_MODEL * D_MODEL * 2);   // 2 MiB
    u16t* wkb = (u16t*)carve((size_t)D_MODEL * D_MODEL * 2);
    u16t* wvb = (u16t*)carve((size_t)D_MODEL * D_MODEL * 2);
    u16t* wob = (u16t*)carve((size_t)D_MODEL * D_MODEL * 2);
    u16t* Qr  = (u16t*)carve((size_t)BATCH * NH * SEQ * HD * 2);   // 8 MiB
    u16t* Kr  = (u16t*)carve((size_t)BATCH * NH * SEQ * HD * 2);
    u16t* Vt  = (u16t*)carve((size_t)BATCH * NH * HD * SEQ * 2);
    u16t* AO  = (u16t*)carve((size_t)NROWS * D_MODEL * 2);     // must follow Vt

    const int xn4 = NROWS * D_MODEL / 4;
    const int wn4 = D_MODEL * D_MODEL / 4;
    cvt_f32_to_bf16<<<(xn4 + 255) / 256, 256, 0, stream>>>((const float4*)x,  (ushort4*)xb,  xn4);
    cvt_f32_to_bf16<<<(wn4 + 255) / 256, 256, 0, stream>>>((const float4*)Wq, (ushort4*)wqb, wn4);
    cvt_f32_to_bf16<<<(wn4 + 255) / 256, 256, 0, stream>>>((const float4*)Wk, (ushort4*)wkb, wn4);
    cvt_f32_to_bf16<<<(wn4 + 255) / 256, 256, 0, stream>>>((const float4*)Wv, (ushort4*)wvb, wn4);
    cvt_f32_to_bf16<<<(wn4 + 255) / 256, 256, 0, stream>>>((const float4*)Wo, (ushort4*)wob, wn4);

    qkv_rope_kernel<<<dim3(32, 8, 3), 256, 0, stream>>>(xb, wqb, wkb, wvb, Qr, Kr, Vt);
    attn_kernel<<<dim3(32, 16), 256, 0, stream>>>(Qr, Kr, Vt, AO);
    oproj_kernel<<<dim3(32, 8), 256, 0, stream>>>(AO, wob, out);
}
